// Genotype2PhenotypeTransformer_83511344103743
// MI455X (gfx1250) — compile-verified, matrix operand order checked by reading, not run on hardware
//
#include <hip/hip_runtime.h>
#include <hip/hip_bf16.h>

// ---------------------------------------------------------------------------
// Problem constants (from reference)
// ---------------------------------------------------------------------------
constexpr int kB    = 16;
constexpr int kNSYS = 1024;
constexpr int kD    = 256;
constexpr int kH    = 4;
constexpr int kL    = 2;
constexpr int kDH   = 64;           // D / H
constexpr int kDFF  = 1024;         // 4 * D
constexpr int kM    = kB * kNSYS;   // 16384 rows
constexpr float kEpsUpd = 0.1f;
constexpr float kEpsSys = 1e-5f;

typedef __attribute__((ext_vector_type(16))) __bf16 v16bf;
typedef __attribute__((ext_vector_type(8)))  float  v8f;

// ---------------------------------------------------------------------------
// bf16 helpers (bit-level, no reliance on bf16 arithmetic support)
// ---------------------------------------------------------------------------
__device__ __forceinline__ __bf16 f2bf(float f) {
    unsigned u = __float_as_uint(f);
    unsigned r = (u + 0x7FFFu + ((u >> 16) & 1u)) >> 16;
    unsigned short s = (unsigned short)r;
    return *reinterpret_cast<__bf16*>(&s);
}
__device__ __forceinline__ float gelu_tanh(float x) {
    float x3 = x * x * x;
    return 0.5f * x * (1.f + tanhf(0.7978845608028654f * (x + 0.044715f * x3)));
}
__device__ __forceinline__ float blockReduceSum256(float v, float* red) {
    int t = threadIdx.x;
    red[t] = v;
    __syncthreads();
    #pragma unroll
    for (int s = 128; s > 0; s >>= 1) {
        if (t < s) red[t] += red[t + s];
        __syncthreads();
    }
    float r = red[0];
    __syncthreads();
    return r;
}

// ---------------------------------------------------------------------------
// fp32 -> bf16 plain conversion
// ---------------------------------------------------------------------------
__global__ void convert_bf16_kernel(const float* __restrict__ src,
                                    __bf16* __restrict__ dst, int n) {
    int i = blockIdx.x * blockDim.x + threadIdx.x;
    if (i < n) dst[i] = f2bf(src[i]);
}

// ---------------------------------------------------------------------------
// fp32 W[K,N] -> bf16 fragment-swizzled Wp.
// Layout: Wp[(((k0s * (N/16) + nt) * 32 + lane) * 16 + e], where for lane,
// e = 2*j+s the element is W[k0s*32 + kb][nt*16 + (lane&15)],
// kb = (j>=4 ? 16:0) + (lane>>4)*8 + 2*(j&3) + s   (WMMA bf16 B layout).
// Each lane's 16 B-operand bf16 are contiguous (32 bytes) -> b128 loads.
// ---------------------------------------------------------------------------
__global__ void convert_swizzleB_kernel(const float* __restrict__ W,
                                        __bf16* __restrict__ Wp,
                                        int N, int total) {
    int i = blockIdx.x * blockDim.x + threadIdx.x;
    if (i >= total) return;
    int e    = i & 15;
    int lane = (i >> 4) & 31;
    int tmp  = i >> 9;
    int NT   = N >> 4;
    int nt   = tmp % NT;
    int k0s  = tmp / NT;
    int j = e >> 1, s = e & 1;
    int kb = ((j >= 4) ? 16 : 0) + ((lane >> 4) * 8) + 2 * (j & 3) + s;
    Wp[i] = f2bf(W[(size_t)(k0s * 32 + kb) * N + (nt * 16 + (lane & 15))]);
}

// ---------------------------------------------------------------------------
// Gather + LayerNorm: y = LN(table[idx[n]] + upd[b, idx[n], :])
// ---------------------------------------------------------------------------
template <bool OUTF>
__global__ __launch_bounds__(256)
void ln_gather_kernel(const float* __restrict__ table,
                      const float* __restrict__ upd,
                      const int*   __restrict__ idx,
                      const float* __restrict__ gamma,
                      const float* __restrict__ beta,
                      float eps,
                      float* __restrict__ outF,
                      __bf16* __restrict__ outB) {
    __shared__ float red[256];
    int b = blockIdx.x / kNSYS;
    int n = blockIdx.x - b * kNSYS;
    int d = threadIdx.x;
    int id = idx[n];
    float x = table[(size_t)id * kD + d] + upd[((size_t)b * kNSYS + id) * kD + d];
    float mu = blockReduceSum256(x, red) * (1.f / kD);
    float dv = x - mu;
    float var = blockReduceSum256(dv * dv, red) * (1.f / kD);
    float y = dv * rsqrtf(var + eps) * gamma[d] + beta[d];
    size_t o = ((size_t)b * kNSYS + n) * kD + d;
    if constexpr (OUTF) outF[o] = y;
    outB[o] = f2bf(y);
}

// ---------------------------------------------------------------------------
// add + LayerNorm: y = LN(a + c)
// XOUT: write y to outF (f32) and outB (bf16).  ACC: acc += y.
// ---------------------------------------------------------------------------
template <bool XOUT, bool ACC>
__global__ __launch_bounds__(256)
void add_ln_kernel(const float* __restrict__ a,
                   const float* __restrict__ c,
                   const float* __restrict__ gamma,
                   const float* __restrict__ beta,
                   float eps,
                   float* __restrict__ outF,
                   __bf16* __restrict__ outB,
                   float* __restrict__ acc) {
    __shared__ float red[256];
    size_t o = (size_t)blockIdx.x * kD + threadIdx.x;
    int d = threadIdx.x;
    float x = a[o] + c[o];
    float mu = blockReduceSum256(x, red) * (1.f / kD);
    float dv = x - mu;
    float var = blockReduceSum256(dv * dv, red) * (1.f / kD);
    float y = dv * rsqrtf(var + eps) * gamma[d] + beta[d];
    if constexpr (XOUT) { outF[o] = y; outB[o] = f2bf(y); }
    if constexpr (ACC)  acc[o] += y;
}

// ---------------------------------------------------------------------------
// bf16 WMMA GEMM: C[M,N] = A[M,K] x W[K,N]
//  - A row-major bf16, staged via LDS (contiguous-K fragments -> ds_load_b128)
//  - W pre-swizzled (convert_swizzleB_kernel) -> direct global b128 frag loads
//  - 256 threads (8 waves), block tile 128x64, each wave 32x32 (2x2 WMMA)
//  - MODE: 0 = store f32, 1 = store bf16, 2 = store bf16(gelu(x))
// M % 128 == 0, N % 64 == 0, K % 32 == 0 at all call sites.
// ---------------------------------------------------------------------------
template <int MODE>
__global__ __launch_bounds__(256)
void gemm_bf16_kernel(const __bf16* __restrict__ A,
                      const __bf16* __restrict__ Wp,
                      float* __restrict__ outF,
                      __bf16* __restrict__ outB,
                      int M, int N, int K) {
    __shared__ __align__(16) __bf16 As[128][40];

    const int m0 = blockIdx.x * 128;
    const int n0 = blockIdx.y * 64;
    const int NT = N >> 4;
    const int t = threadIdx.x;
    const int wave = t >> 5, lane = t & 31;
    const int lane15 = lane & 15;
    const int hi8 = (lane >> 4) * 8;
    const int wm = (wave & 3) * 32;
    const int wn = (wave >> 2) * 32;

    const int ar = t >> 1, ac0 = (t & 1) * 16;  // A staging assignment

    v8f acc[2][2] = {};

    for (int k0 = 0; k0 < K; k0 += 32) {
        {   // stage A tile 128x32 (two 16B vector copies per thread)
            const float4* src = reinterpret_cast<const float4*>(
                A + (size_t)(m0 + ar) * K + k0 + ac0);
            float4* dst = reinterpret_cast<float4*>(&As[ar][ac0]);
            dst[0] = src[0]; dst[1] = src[1];
        }
        if (k0 + 32 < K) {  // prefetch next A tile (global_prefetch_b8)
            __builtin_prefetch(A + (size_t)(m0 + ar) * K + k0 + 32 + ac0, 0, 3);
        }
        __syncthreads();

        // B fragments: contiguous 32B per lane from swizzled weights
        const int k0s = k0 >> 5;
        v16bf fb[2];
        #pragma unroll
        for (int i = 0; i < 2; ++i) {
            int nt = (n0 + wn + i * 16) >> 4;
            fb[i] = *reinterpret_cast<const v16bf*>(
                Wp + (((size_t)k0s * NT + nt) * 32 + lane) * 16);
        }
        // A fragments: contiguous K runs in LDS
        v16bf fa[2];
        #pragma unroll
        for (int i = 0; i < 2; ++i) {
            int mB = wm + i * 16 + lane15;
            #pragma unroll
            for (int j = 0; j < 8; ++j) {
                int kb = ((j >= 4) ? 16 : 0) + hi8 + 2 * (j & 3);
                fa[i][2 * j]     = As[mB][kb];
                fa[i][2 * j + 1] = As[mB][kb + 1];
            }
        }
        #pragma unroll
        for (int i = 0; i < 2; ++i)
            #pragma unroll
            for (int j = 0; j < 2; ++j)
                acc[i][j] = __builtin_amdgcn_wmma_f32_16x16x32_bf16(
                    false, fa[i], false, fb[j], (short)0, acc[i][j], false, false);
        __syncthreads();
    }

    #pragma unroll
    for (int i = 0; i < 2; ++i)
        #pragma unroll
        for (int j = 0; j < 2; ++j)
            #pragma unroll
            for (int r = 0; r < 8; ++r) {
                int row = m0 + wm + i * 16 + r + hi8;
                int col = n0 + wn + j * 16 + lane15;
                size_t o = (size_t)row * N + col;
                float v = acc[i][j][r];
                if constexpr (MODE == 0)      outF[o] = v;
                else if constexpr (MODE == 1) outB[o] = f2bf(v);
                else                          outB[o] = f2bf(gelu_tanh(v));
            }
}

// ---------------------------------------------------------------------------
// Flash-style attention: one block (128 thr / 4 waves) per (b, h, 32-q tile)
// S = (Q K^T)/8 masked; online softmax; ctx = P V; ctx -> bf16
// V is staged TRANSPOSED in LDS so the P*V B-operand is contiguous.
// ---------------------------------------------------------------------------
__global__ __launch_bounds__(128)
void attention_kernel(const __bf16* __restrict__ Qb,
                      const __bf16* __restrict__ Kb,
                      const __bf16* __restrict__ Vb,
                      const int*    __restrict__ maskl,
                      __bf16* __restrict__ ctxb) {
    __shared__ __align__(16) __bf16 Qs[32][72];
    __shared__ __align__(16) __bf16 Ks[32][72];
    __shared__ __align__(16) __bf16 Vts[64][40];   // [d][kv] (transposed)
    __shared__ float  Ss[32][33];
    __shared__ __align__(16) __bf16 Ps[32][40];
    __shared__ float rowM[32], rowL[32], rowAlpha[32];

    const int q0 = blockIdx.x * 32;
    const int h  = blockIdx.y;
    const int b  = blockIdx.z;
    const int t = threadIdx.x;
    const int wave = t >> 5, lane = t & 31;
    const int lane15 = lane & 15;
    const int hi8 = (lane >> 4) * 8;
    const int qh = (wave & 1) * 16;   // q half owned by wave (S and ctx)
    const int kh = (wave >> 1) * 16;  // k half for S
    const int dh = (wave >> 1) * 32;  // d half for ctx

    const int lr = t >> 2, lc0 = (t & 3) * 16;  // staging assignment

    // load Q tile (32 x 64)
    {
        const float4* src = reinterpret_cast<const float4*>(
            Qb + ((size_t)(b * kNSYS + q0 + lr)) * kD + h * kDH + lc0);
        float4* dst = reinterpret_cast<float4*>(&Qs[lr][lc0]);
        dst[0] = src[0]; dst[1] = src[1];
    }
    if (t < 32) { rowM[t] = -3.0e38f; rowL[t] = 0.f; }

    v8f accC[2] = {};

    for (int kt = 0; kt < kNSYS; kt += 32) {
        {   // stage K chunk row-major; V chunk transposed
            const float4* sk = reinterpret_cast<const float4*>(
                Kb + ((size_t)(b * kNSYS + kt + lr)) * kD + h * kDH + lc0);
            float4* dk = reinterpret_cast<float4*>(&Ks[lr][lc0]);
            dk[0] = sk[0]; dk[1] = sk[1];

            const float4* sv = reinterpret_cast<const float4*>(
                Vb + ((size_t)(b * kNSYS + kt + lr)) * kD + h * kDH + lc0);
            __bf16 tmpv[16];
            *reinterpret_cast<float4*>(&tmpv[0]) = sv[0];
            *reinterpret_cast<float4*>(&tmpv[8]) = sv[1];
            #pragma unroll
            for (int x = 0; x < 16; ++x) Vts[lc0 + x][lr] = tmpv[x];
        }
        __syncthreads();

        // ---- S tile: each wave one 16x16 quadrant, K-dim = DH = 2x32 ----
        v8f sacc = {};
        #pragma unroll
        for (int ks = 0; ks < 2; ++ks) {
            int d0 = ks * 32;
            v16bf fq, fk;
            #pragma unroll
            for (int j = 0; j < 8; ++j) {
                int kb = ((j >= 4) ? 16 : 0) + hi8 + 2 * (j & 3);
                fq[2 * j]     = Qs[qh + lane15][d0 + kb];
                fq[2 * j + 1] = Qs[qh + lane15][d0 + kb + 1];
                // B operand = K^T : B[kb][n] = K[n][kb] (row reads, contiguous)
                fk[2 * j]     = Ks[kh + lane15][d0 + kb];
                fk[2 * j + 1] = Ks[kh + lane15][d0 + kb + 1];
            }
            sacc = __builtin_amdgcn_wmma_f32_16x16x32_bf16(
                false, fq, false, fk, (short)0, sacc, false, false);
        }
        #pragma unroll
        for (int r = 0; r < 8; ++r) {
            int ql = qh + r + hi8;
            int kl = kh + lane15;
            int mv = maskl[(size_t)(q0 + ql) * kNSYS + (kt + kl)];
            Ss[ql][kl] = (mv > 0) ? sacc[r] * 0.125f : -1.0e9f;
        }
        __syncthreads();

        // ---- online softmax: thread t < 32 owns row t ----
        if (t < 32) {
            float mOld = rowM[t];
            float cmax = -3.0e38f;
            #pragma unroll 8
            for (int j = 0; j < 32; ++j) cmax = fmaxf(cmax, Ss[t][j]);
            float mNew = fmaxf(mOld, cmax);
            float alpha = expf(mOld - mNew);
            float psum = 0.f;
            #pragma unroll 8
            for (int j = 0; j < 32; ++j) {
                float p = expf(Ss[t][j] - mNew);
                Ps[t][j] = f2bf(p);
                psum += p;
            }
            rowL[t] = rowL[t] * alpha + psum;
            rowM[t] = mNew;
            rowAlpha[t] = alpha;
        }
        __syncthreads();

        // ---- rescale ctx accumulators per-row, then ctx += P x V ----
        #pragma unroll
        for (int r = 0; r < 8; ++r) {
            float a = rowAlpha[qh + r + hi8];
            accC[0][r] *= a;
            accC[1][r] *= a;
        }
        v16bf fp;
        #pragma unroll
        for (int j = 0; j < 8; ++j) {
            int kb = ((j >= 4) ? 16 : 0) + hi8 + 2 * (j & 3);
            fp[2 * j]     = Ps[qh + lane15][kb];
            fp[2 * j + 1] = Ps[qh + lane15][kb + 1];
        }
        #pragma unroll
        for (int tn = 0; tn < 2; ++tn) {
            int nB = dh + tn * 16 + lane15;
            v16bf fv;
            #pragma unroll
            for (int j = 0; j < 8; ++j) {
                int kb = ((j >= 4) ? 16 : 0) + hi8 + 2 * (j & 3);
                // B operand (k=kv, n=d) = V[kv][d] = Vts[d][kv] (contiguous)
                fv[2 * j]     = Vts[nB][kb];
                fv[2 * j + 1] = Vts[nB][kb + 1];
            }
            accC[tn] = __builtin_amdgcn_wmma_f32_16x16x32_bf16(
                false, fp, false, fv, (short)0, accC[tn], false, false);
        }
        __syncthreads();
    }

    // ---- finalize: ctx / rowL, store bf16 ----
    #pragma unroll
    for (int tn = 0; tn < 2; ++tn)
        #pragma unroll
        for (int r = 0; r < 8; ++r) {
            int ql = qh + r + hi8;
            int dc = dh + tn * 16 + lane15;
            float v = accC[tn][r] / rowL[ql];
            ctxb[((size_t)(b * kNSYS + q0 + ql)) * kD + h * kDH + dc] = f2bf(v);
        }
}

// ---------------------------------------------------------------------------
// out = system_embedding + upd
// ---------------------------------------------------------------------------
__global__ void final_add_kernel(const float* __restrict__ se,
                                 const float* __restrict__ upd,
                                 float* __restrict__ out, int n) {
    int i = blockIdx.x * blockDim.x + threadIdx.x;
    if (i < n) out[i] = se[i] + upd[i];
}

// ---------------------------------------------------------------------------
// Host orchestration
// ---------------------------------------------------------------------------
extern "C" void kernel_launch(void* const* d_in, const int* in_sizes, int n_in,
                              void* d_out, int out_size, void* d_ws, size_t ws_size,
                              hipStream_t stream) {
    const float* sys_emb = (const float*)d_in[0];
    const float* upd0    = (const float*)d_in[1];
    const float* table   = (const float*)d_in[2];
    const float* Wq = (const float*)d_in[3];
    const float* Wk = (const float*)d_in[4];
    const float* Wv = (const float*)d_in[5];
    const float* Wo = (const float*)d_in[6];
    const float* W1 = (const float*)d_in[7];
    const float* W2 = (const float*)d_in[8];
    const float* sys_g = (const float*)d_in[9];
    const float* sys_b = (const float*)d_in[10];
    const float* in_g  = (const float*)d_in[11];
    const float* in_b  = (const float*)d_in[12];
    const float* out_g = (const float*)d_in[13];
    const float* out_b = (const float*)d_in[14];
    const int* qidx = (const int*)d_in[15];
    const int* kidx = (const int*)d_in[16];
    const int* mask = (const int*)d_in[17];
    float* out = (float*)d_out;

    const size_t elemMD = (size_t)kM * kD;      // 4,194,304
    const size_t elemMF = (size_t)kM * kDFF;    // 16,777,216
    const size_t wDD = (size_t)kD * kD;         // one layer's DxD matrix
    const size_t wDF = (size_t)kD * kDFF;       // one layer's DxDFF matrix

    char* ws = (char*)d_ws;
    size_t off = 0;
    auto take = [&](size_t bytes) -> char* {
        char* p = ws + off;
        off += (bytes + 255) & ~(size_t)255;
        return p;
    };
    float*  updW = (float*)take(elemMD * 4);
    float*  qf   = (float*)take(elemMD * 4);
    __bf16* qb   = (__bf16*)take(elemMD * 2);
    __bf16* kb   = (__bf16*)take(elemMD * 2);
    __bf16* Qb   = (__bf16*)take(elemMD * 2);
    __bf16* Kbuf = (__bf16*)take(elemMD * 2);
    __bf16* Vb   = (__bf16*)take(elemMD * 2);
    __bf16* ctxb = (__bf16*)take(elemMD * 2);
    float*  t0   = (float*)take(elemMD * 4);
    float*  xf   = (float*)take(elemMD * 4);
    __bf16* xb   = (__bf16*)take(elemMD * 2);
    __bf16* gb   = (__bf16*)take(elemMF * 2);
    __bf16* wqb  = (__bf16*)take(kL * wDD * 2);
    __bf16* wkb  = (__bf16*)take(kL * wDD * 2);
    __bf16* wvb  = (__bf16*)take(kL * wDD * 2);
    __bf16* wob  = (__bf16*)take(kL * wDD * 2);
    __bf16* w1b  = (__bf16*)take(kL * wDF * 2);
    __bf16* w2b  = (__bf16*)take(kL * wDF * 2);

    // upd = update_tensor (copy; inputs must not be mutated)
    hipMemcpyAsync(updW, upd0, elemMD * 4, hipMemcpyDeviceToDevice, stream);

    // weights -> bf16, fragment-swizzled, per layer matrix
    auto swz = [&](const float* s, __bf16* d, int N, size_t n) {
        convert_swizzleB_kernel<<<dim3((unsigned)((n + 255) / 256)), 256, 0, stream>>>(
            s, d, N, (int)n);
    };
    for (int l = 0; l < kL; ++l) {
        swz(Wq + l * wDD, wqb + l * wDD, kD,   wDD);
        swz(Wk + l * wDD, wkb + l * wDD, kD,   wDD);
        swz(Wv + l * wDD, wvb + l * wDD, kD,   wDD);
        swz(Wo + l * wDD, wob + l * wDD, kD,   wDD);
        swz(W1 + l * wDF, w1b + l * wDF, kDFF, wDF);
        swz(W2 + l * wDF, w2b + l * wDF, kD,   wDF);
    }

    const dim3 gemmBlk(256);
    const dim3 lnGrid(kM);

    for (int l = 0; l < kL; ++l) {
        const __bf16* wq_l = wqb + (size_t)l * wDD;
        const __bf16* wk_l = wkb + (size_t)l * wDD;
        const __bf16* wv_l = wvb + (size_t)l * wDD;
        const __bf16* wo_l = wob + (size_t)l * wDD;
        const __bf16* w1_l = w1b + (size_t)l * wDF;
        const __bf16* w2_l = w2b + (size_t)l * wDF;
        const int* qi_l = qidx + (size_t)l * kNSYS;
        const int* ki_l = kidx + (size_t)l * kNSYS;
        const int* mk_l = mask + (size_t)l * kNSYS * kNSYS;

        // q = LN(table[qi] + upd[:,qi,:]) ; k likewise (bf16 only)
        ln_gather_kernel<true><<<lnGrid, 256, 0, stream>>>(
            table, updW, qi_l, sys_g, sys_b, kEpsSys, qf, qb);
        ln_gather_kernel<false><<<lnGrid, 256, 0, stream>>>(
            table, updW, ki_l, sys_g, sys_b, kEpsSys, nullptr, kb);

        // Q/K/V projections (bf16 out)
        dim3 gP(kM / 128, kD / 64);
        gemm_bf16_kernel<1><<<gP, gemmBlk, 0, stream>>>(qb, wq_l, nullptr, Qb, kM, kD, kD);
        gemm_bf16_kernel<1><<<gP, gemmBlk, 0, stream>>>(kb, wk_l, nullptr, Kbuf, kM, kD, kD);
        gemm_bf16_kernel<1><<<gP, gemmBlk, 0, stream>>>(kb, wv_l, nullptr, Vb, kM, kD, kD);

        // attention
        dim3 gA(kNSYS / 32, kH, kB);
        attention_kernel<<<gA, dim3(128), 0, stream>>>(Qb, Kbuf, Vb, mk_l, ctxb);

        // O = ctx @ Wo (f32) ; x = LN(q + O)
        gemm_bf16_kernel<0><<<gP, gemmBlk, 0, stream>>>(ctxb, wo_l, t0, nullptr, kM, kD, kD);
        add_ln_kernel<true, false><<<lnGrid, 256, 0, stream>>>(
            qf, t0, in_g, in_b, kEpsUpd, xf, xb, nullptr);

        // FFN: g = gelu(x @ W1) [bf16], F2 = g @ W2 [f32]
        dim3 g1(kM / 128, kDFF / 64);
        gemm_bf16_kernel<2><<<g1, gemmBlk, 0, stream>>>(xb, w1_l, nullptr, gb, kM, kDFF, kD);
        gemm_bf16_kernel<0><<<gP, gemmBlk, 0, stream>>>(gb, w2_l, t0, nullptr, kM, kD, kDFF);

        // upd += LN(x + F2)
        add_ln_kernel<false, true><<<lnGrid, 256, 0, stream>>>(
            xf, t0, out_g, out_b, kEpsUpd, nullptr, nullptr, updW);
    }

    final_add_kernel<<<dim3((unsigned)((elemMD + 255) / 256)), 256, 0, stream>>>(
        sys_emb, updW, out, (int)elemMD);
}
